// LocalMHA_13589276524640
// MI455X (gfx1250) — matrix operand order repaired and checked by reading, not compiled, not run on hardware
//
#include <hip/hip_runtime.h>
#include <hip/hip_bf16.h>

// ---------------------------------------------------------------------------
// LocalMHA for MI455X (gfx1250, wave32, WMMA 16x16x32 f16)
// ---------------------------------------------------------------------------

typedef _Float16 v16h __attribute__((ext_vector_type(16)));
typedef _Float16 v8h  __attribute__((ext_vector_type(8)));
typedef float    v8f  __attribute__((ext_vector_type(8)));
typedef int      i4v  __attribute__((ext_vector_type(4)));

#define B_    4
#define C_    1024
#define T_    8192
#define H3_   3072
#define HEADS 16
#define DHEAD 64
#define WIN   32
#define NWIN  256          // T_/WIN
#define M_    (B_*T_)      // 32768 rows

// gfx1250 async global->LDS copy path (ASYNCcnt tracked), guarded so the file
// still compiles with toolchains that don't declare the builtins.
#if defined(__HIP_DEVICE_COMPILE__) &&                                         \
    __has_builtin(__builtin_amdgcn_global_load_async_to_lds_b128) &&           \
    __has_builtin(__builtin_amdgcn_s_wait_asynccnt)
#define HAVE_ASYNC_LDS 1
#else
#define HAVE_ASYNC_LDS 0
#endif

// ---------------------------------------------------------------------------
// helpers
// ---------------------------------------------------------------------------
__device__ inline v8f zero8() {
    v8f z;
#pragma unroll
    for (int i = 0; i < 8; ++i) z[i] = 0.0f;
    return z;
}

__device__ inline v16h join16(v8h lo, v8h hi) {
    return __builtin_shufflevector(lo, hi, 0,1,2,3,4,5,6,7,8,9,10,11,12,13,14,15);
}

// A fragment: 16xK tile (row-major, row stride ldk halves), K-step of 32.
// Wave32 layout: lane<16 -> row=lane, K in {0..7,16..23}; lane>=16 -> row=lane-16,
// K in {8..15,24..31}.  Two contiguous 16B chunks -> 2x ds_load_b128.
__device__ inline v16h load_frag_a(const _Float16* base, int lane, int ldk) {
    const int row = lane & 15;
    const int kb  = (lane < 16) ? 0 : 8;
    const _Float16* p = base + row * ldk + kb;
    v8h lo = *(const v8h*)p;
    v8h hi = *(const v8h*)(p + 16);
    return join16(lo, hi);
}

// B fragment: 32x16 (KxN) from N-major storage Bt[N][K] (row stride ldk halves).
// lane<16 -> col=lane, K in {0..7,16..23} ; lane>=16 -> col=lane-16, K in {8..15,24..31}.
__device__ inline v16h load_frag_b(const _Float16* baseT, int lane, int ldk) {
    const int n   = lane & 15;
    const int klo = (lane < 16) ? 0 : 8;
    const _Float16* p = baseT + n * ldk + klo;
    v8h lo = *(const v8h*)p;
    v8h hi = *(const v8h*)(p + 16);
    return join16(lo, hi);
}

__device__ inline v8f wmma32(v16h a, v16h b, v8f c) {
    return __builtin_amdgcn_wmma_f32_16x16x32_f16(false, a, false, b,
                                                  (short)0, c, false, false);
}

#if HAVE_ASYNC_LDS
// signature (from hipcc diagnostic): param0 = global int4*, param1 = LDS int4*
__device__ inline void async_cp_b128(const _Float16* g, _Float16* l) {
    __builtin_amdgcn_global_load_async_to_lds_b128(
        (__attribute__((address_space(1))) i4v*)(_Float16*)g,
        (__attribute__((address_space(3))) i4v*)l, 0, 0);
}
#endif

// ---------------------------------------------------------------------------
// K0: fp32 -> fp16 weight conversion
// ---------------------------------------------------------------------------
__global__ __launch_bounds__(256) void f32_to_f16(const float* __restrict__ src,
                                                  _Float16* __restrict__ dst, int n) {
    int i = blockIdx.x * blockDim.x + threadIdx.x;
    if (i < n) dst[i] = (_Float16)src[i];
}

// ---------------------------------------------------------------------------
// K1: LayerNorm over C with implicit [B,C,T] -> [B,T,C] transpose. Output f16.
// One block = 8 tokens. LDS tile padded to 1025 floats/row (conflict-free reads).
// ---------------------------------------------------------------------------
__global__ __launch_bounds__(256) void ln_kernel(const float* __restrict__ x,
                                                 const float* __restrict__ gamma,
                                                 const float* __restrict__ beta,
                                                 _Float16* __restrict__ xn) {
    __shared__ float tile[8][1025];
    const int tpb = T_ / 8;
    const int b   = blockIdx.x / tpb;
    const int t0  = (blockIdx.x % tpb) * 8;
    const int tid = threadIdx.x;
    const int tt  = tid & 7;        // token within tile
    const int cb  = tid >> 3;       // 0..31 channel group

    const size_t xbase = (size_t)b * C_ * T_ + (size_t)t0 + tt;
#pragma unroll 4
    for (int s = 0; s < 32; ++s) {
        const int c = cb + 32 * s;
        tile[tt][c] = x[xbase + (size_t)c * T_];
    }
    __syncthreads();

    const int wv   = tid >> 5;      // wave -> row
    const int lane = tid & 31;
    float s1 = 0.f, s2 = 0.f;
#pragma unroll 8
    for (int i = 0; i < 32; ++i) {
        float v = tile[wv][lane + 32 * i];
        s1 += v; s2 += v * v;
    }
#pragma unroll
    for (int off = 16; off >= 1; off >>= 1) {
        s1 += __shfl_xor(s1, off, 32);
        s2 += __shfl_xor(s2, off, 32);
    }
    const float mu   = s1 * (1.0f / C_);
    const float var  = s2 * (1.0f / C_) - mu * mu;
    const float rstd = rsqrtf(var + 1e-5f);

    const size_t rowOut = ((size_t)(b * T_) + t0 + wv) * C_;
#pragma unroll 8
    for (int i = 0; i < 32; ++i) {
        const int c = lane + 32 * i;
        const float v = (tile[wv][c] - mu) * rstd * gamma[c] + beta[c];
        xn[rowOut + c] = (_Float16)v;
    }
}

// ---------------------------------------------------------------------------
// K2/K4: f16 WMMA GEMM  D[M,N] = A[M,K] * B[K,N]
// Block tile 128x128, 8 waves (4Mx2N), wave tile 32x64 -> 8 accumulators.
// Double-buffered LDS; A staged via async global->LDS b128 when available;
// B transposed into Bt[N][K] so B fragments are contiguous per lane.
// ---------------------------------------------------------------------------
template <typename OutT>
__global__ __launch_bounds__(256) void gemm_f16(const _Float16* __restrict__ A,
                                                const _Float16* __restrict__ Bw,
                                                OutT* __restrict__ D,
                                                int M, int N, int K) {
    __shared__ _Float16 As[2][128][32];
    __shared__ _Float16 Bt[2][128][32];

    const int tid  = threadIdx.x;
    const int lane = tid & 31;
    const int wv   = tid >> 5;
    const int mW   = (wv & 3) * 32;     // wave M offset in block tile
    const int nW   = (wv >> 2) * 64;    // wave N offset in block tile
    const int blockM = blockIdx.y * 128;
    const int blockN = blockIdx.x * 128;

    // staging decomposition (256 threads)
    const int arow = tid >> 1;          // A row 0..127
    const int ach  = (tid & 1) * 16;    // A k-chunk 0 / 16
    const int kr   = tid >> 3;          // B k-row 0..31
    const int ncb  = (tid & 7) * 16;    // B n-chunk 0..112

    auto stage = [&](int buf, int k0) {
        const _Float16* ga = A + (size_t)(blockM + arow) * K + k0 + ach;
#if HAVE_ASYNC_LDS
        async_cp_b128(ga,     &As[buf][arow][ach]);
        async_cp_b128(ga + 8, &As[buf][arow][ach + 8]);
#else
        *(v8h*)&As[buf][arow][ach]     = *(const v8h*)ga;
        *(v8h*)&As[buf][arow][ach + 8] = *(const v8h*)(ga + 8);
#endif
        const _Float16* gb = Bw + (size_t)(k0 + kr) * N + blockN + ncb;
        v8h b0 = *(const v8h*)gb;
        v8h b1 = *(const v8h*)(gb + 8);
#pragma unroll
        for (int e = 0; e < 8; ++e) {
            Bt[buf][ncb + e][kr]     = b0[e];
            Bt[buf][ncb + 8 + e][kr] = b1[e];
        }
    };

    v8f acc[2][4];
#pragma unroll
    for (int mi = 0; mi < 2; ++mi)
#pragma unroll
        for (int j = 0; j < 4; ++j) acc[mi][j] = zero8();

    stage(0, 0);
#if HAVE_ASYNC_LDS
    __builtin_amdgcn_s_wait_asynccnt(0);
#endif
    __syncthreads();

    const int nIter = K / 32;
    for (int it = 0; it < nIter; ++it) {
        const int cur = it & 1;
        if (it + 1 < nIter) {
            stage(cur ^ 1, (it + 1) * 32);
            if (it + 2 < nIter) {   // L2 prefetch two k-tiles ahead
                __builtin_prefetch(A + (size_t)(blockM + arow) * K + (it + 2) * 32 + ach, 0, 0);
                __builtin_prefetch(Bw + (size_t)((it + 2) * 32 + kr) * N + blockN + ncb, 0, 0);
            }
        }

        v16h aF0 = load_frag_a(&As[cur][mW][0],      lane, 32);
        v16h aF1 = load_frag_a(&As[cur][mW + 16][0], lane, 32);
#pragma unroll
        for (int j = 0; j < 4; ++j) {
            v16h bF = load_frag_b(&Bt[cur][nW + j * 16][0], lane, 32);
            acc[0][j] = wmma32(aF0, bF, acc[0][j]);
            acc[1][j] = wmma32(aF1, bF, acc[1][j]);
        }
#if HAVE_ASYNC_LDS
        __builtin_amdgcn_s_wait_asynccnt(0);
#endif
        __syncthreads();
    }

    // ---- store D (C/D layout: col = lane&15, row = r + 8*(lane>=16)) ----
    const int n  = lane & 15;
    const int mo = (lane < 16) ? 0 : 8;
#pragma unroll
    for (int mi = 0; mi < 2; ++mi)
#pragma unroll
        for (int j = 0; j < 4; ++j)
#pragma unroll
            for (int r = 0; r < 8; ++r) {
                const size_t row = (size_t)(blockM + mW + mi * 16 + mo + r);
                D[row * N + blockN + nW + j * 16 + n] = (OutT)acc[mi][j][r];
            }
}

// ---------------------------------------------------------------------------
// K3: windowed attention. 1 wave per (b, head, window). RoPE fused into the
// global->LDS load. scores = q*k^T (8 WMMA), softmax in f32, out = P*v (8 WMMA).
// ---------------------------------------------------------------------------
__global__ __launch_bounds__(64) void attn_kernel(const _Float16* __restrict__ qkv,
                                                  _Float16* __restrict__ aout) {
    __shared__ _Float16 qs[2][32][64];
    __shared__ _Float16 ks[2][32][64];
    __shared__ _Float16 vts[2][64][32];   // v transposed: [d][n]
    __shared__ float    sc[2][32][32];
    __shared__ _Float16 at[2][32][32];

    const int lane = threadIdx.x & 31;
    const int wv   = threadIdx.x >> 5;
    const int g    = blockIdx.x;
    const int hq   = g & 7;
    const int bw   = g >> 3;
    const int w    = bw & (NWIN - 1);
    const int b    = bw >> 8;
    const int h    = hq * 2 + wv;
    const int t0   = w * WIN;

    _Float16 (*qw)[64]  = qs[wv];
    _Float16 (*kw)[64]  = ks[wv];
    _Float16 (*vtw)[32] = vts[wv];
    float    (*sw)[32]  = sc[wv];
    _Float16 (*atw)[32] = at[wv];

    // ---- load q,k,v rows (lane = window position) + RoPE on q,k ----
    const int r = lane;
    const size_t rowBase = (size_t)(b * T_ + t0 + r) * H3_;
    const _Float16* qg = qkv + rowBase + h * DHEAD;
    const _Float16* kg = qg + C_;       // k at column offset C
    const _Float16* vg = qg + 2 * C_;   // v at column offset 2C
    const float rr = (float)r;

#pragma unroll
    for (int c2 = 0; c2 < 4; ++c2) {
        v8h q1 = *(const v8h*)(qg + c2 * 8), q2 = *(const v8h*)(qg + c2 * 8 + 32);
        v8h k1 = *(const v8h*)(kg + c2 * 8), k2 = *(const v8h*)(kg + c2 * 8 + 32);
        v8h qo1, qo2, ko1, ko2;
#pragma unroll
        for (int e = 0; e < 8; ++e) {
            const int j = c2 * 8 + e;
            // inv_freq = 10000^(-j/32) = exp2(-j * log2(10000)/32)
            const float ang = rr * exp2f(-0.41524101f * (float)j);
            const float cs = __cosf(ang), sn = __sinf(ang);
            const float a1 = (float)q1[e], a2 = (float)q2[e];
            qo1[e] = (_Float16)(a1 * cs - a2 * sn);
            qo2[e] = (_Float16)(a2 * cs + a1 * sn);
            const float b1 = (float)k1[e], b2 = (float)k2[e];
            ko1[e] = (_Float16)(b1 * cs - b2 * sn);
            ko2[e] = (_Float16)(b2 * cs + b1 * sn);
        }
        *(v8h*)(&qw[r][c2 * 8])      = qo1;
        *(v8h*)(&qw[r][c2 * 8 + 32]) = qo2;
        *(v8h*)(&kw[r][c2 * 8])      = ko1;
        *(v8h*)(&kw[r][c2 * 8 + 32]) = ko2;
    }
#pragma unroll
    for (int c = 0; c < 8; ++c) {
        v8h vv = *(const v8h*)(vg + c * 8);
#pragma unroll
        for (int e = 0; e < 8; ++e) vtw[c * 8 + e][r] = vv[e];
    }
    __syncthreads();

    // ---- scores = q * k^T  (M=32, N=32, K=64) ----
    v8f acc[2][2];
#pragma unroll
    for (int mi = 0; mi < 2; ++mi)
#pragma unroll
        for (int ni = 0; ni < 2; ++ni) acc[mi][ni] = zero8();

#pragma unroll
    for (int kk = 0; kk < 64; kk += 32) {
        v16h aF[2], bF[2];
#pragma unroll
        for (int mi = 0; mi < 2; ++mi) aF[mi] = load_frag_a(&qw[mi * 16][0] + kk, lane, 64);
#pragma unroll
        for (int ni = 0; ni < 2; ++ni) bF[ni] = load_frag_b(&kw[ni * 16][0] + kk, lane, 64);
#pragma unroll
        for (int mi = 0; mi < 2; ++mi)
#pragma unroll
            for (int ni = 0; ni < 2; ++ni)
                acc[mi][ni] = wmma32(aF[mi], bF[ni], acc[mi][ni]);
    }

    const int nC = lane & 15;
    const int mo = (lane < 16) ? 0 : 8;
#pragma unroll
    for (int mi = 0; mi < 2; ++mi)
#pragma unroll
        for (int ni = 0; ni < 2; ++ni)
#pragma unroll
            for (int r8 = 0; r8 < 8; ++r8)
                sw[mi * 16 + mo + r8][ni * 16 + nC] = acc[mi][ni][r8] * 0.125f;
    __syncthreads();

    // ---- softmax: one lane per row ----
    {
        const int m = lane;
        float mx = -1e30f;
#pragma unroll
        for (int j = 0; j < 32; ++j) mx = fmaxf(mx, sw[m][j]);
        float e[32], sum = 0.f;
#pragma unroll
        for (int j = 0; j < 32; ++j) { e[j] = __expf(sw[m][j] - mx); sum += e[j]; }
        const float inv = 1.0f / sum;
#pragma unroll
        for (int j = 0; j < 32; ++j) atw[m][j] = (_Float16)(e[j] * inv);
    }
    __syncthreads();

    // ---- out = P * v  (M=32, N=64, K=32) ----
    v8f acc2[2][4];
#pragma unroll
    for (int mi = 0; mi < 2; ++mi)
#pragma unroll
        for (int nj = 0; nj < 4; ++nj) acc2[mi][nj] = zero8();

    v16h aA[2];
    aA[0] = load_frag_a(&atw[0][0],  lane, 32);
    aA[1] = load_frag_a(&atw[16][0], lane, 32);
#pragma unroll
    for (int nj = 0; nj < 4; ++nj) {
        v16h bB = load_frag_b(&vtw[nj * 16][0], lane, 32);
        acc2[0][nj] = wmma32(aA[0], bB, acc2[0][nj]);
        acc2[1][nj] = wmma32(aA[1], bB, acc2[1][nj]);
    }

#pragma unroll
    for (int mi = 0; mi < 2; ++mi)
#pragma unroll
        for (int nj = 0; nj < 4; ++nj)
#pragma unroll
            for (int r8 = 0; r8 < 8; ++r8) {
                const size_t row = (size_t)(b * T_ + t0 + mi * 16 + mo + r8);
                aout[row * C_ + h * DHEAD + nj * 16 + nC] = (_Float16)acc2[mi][nj][r8];
            }
}

// ---------------------------------------------------------------------------
// K5: [B,T,C] f32 -> [B,C,T] + residual via 32x33 LDS tile
// ---------------------------------------------------------------------------
__global__ __launch_bounds__(256) void transpose_add(const float* __restrict__ tmp,
                                                     const float* __restrict__ x,
                                                     float* __restrict__ out) {
    __shared__ float tile[32][33];
    const int b  = blockIdx.z;
    const int t0 = blockIdx.x * 32;
    const int c0 = blockIdx.y * 32;
    const int j  = threadIdx.x & 31;
    const int i0 = threadIdx.x >> 5;

#pragma unroll
    for (int i = i0; i < 32; i += 8)
        tile[i][j] = tmp[((size_t)(b * T_ + t0 + i)) * C_ + c0 + j];
    __syncthreads();
#pragma unroll
    for (int i = i0; i < 32; i += 8) {
        const size_t o = (size_t)b * C_ * T_ + (size_t)(c0 + i) * T_ + t0 + j;
        out[o] = tile[j][i] + x[o];
    }
}

// ---------------------------------------------------------------------------
// launch
// ---------------------------------------------------------------------------
extern "C" void kernel_launch(void* const* d_in, const int* in_sizes, int n_in,
                              void* d_out, int out_size, void* d_ws, size_t ws_size,
                              hipStream_t stream) {
    (void)in_sizes; (void)n_in; (void)out_size; (void)ws_size;

    const float* x     = (const float*)d_in[0];
    const float* gamma = (const float*)d_in[1];
    const float* beta  = (const float*)d_in[2];
    const float* Wqkv  = (const float*)d_in[3];
    const float* Wout  = (const float*)d_in[4];
    float* out = (float*)d_out;

    char* ws = (char*)d_ws;
    size_t off = 0;
    _Float16* WqkvH = (_Float16*)(ws + off); off += (size_t)C_ * H3_ * 2;
    _Float16* WoutH = (_Float16*)(ws + off); off += (size_t)C_ * C_ * 2;
    _Float16* xn    = (_Float16*)(ws + off); off += (size_t)M_ * C_ * 2;  // also attn-out
    char* big = ws + off;                       // qkv (f16), later out-GEMM temp (f32)
    _Float16* qkv = (_Float16*)big;
    float*    tmp = (float*)big;

    { const int n = C_ * H3_; f32_to_f16<<<(n + 255) / 256, 256, 0, stream>>>(Wqkv, WqkvH, n); }
    { const int n = C_ * C_;  f32_to_f16<<<(n + 255) / 256, 256, 0, stream>>>(Wout, WoutH, n); }

    ln_kernel<<<B_ * (T_ / 8), 256, 0, stream>>>(x, gamma, beta, xn);

    // QKV projection: [32768 x 1024] x [1024 x 3072]
    gemm_f16<_Float16><<<dim3(H3_ / 128, M_ / 128), 256, 0, stream>>>(xn, WqkvH, qkv, M_, H3_, C_);

    // windowed attention (overwrites xn with attention output)
    attn_kernel<<<B_ * NWIN * 8, 64, 0, stream>>>(qkv, xn);

    // output projection: [32768 x 1024] x [1024 x 1024] -> f32 temp (reuses qkv space)
    gemm_f16<float><<<dim3(C_ / 128, M_ / 128), 256, 0, stream>>>(xn, WoutH, tmp, M_, C_, C_);

    // transpose back + residual
    transpose_add<<<dim3(T_ / 32, C_ / 32, B_), 256, 0, stream>>>(tmp, x, out);
}
